// ASTERIS4_59957743452866
// MI455X (gfx1250) — matrix operand order repaired and checked by reading, compiled, not run on hardware
//
#include <hip/hip_runtime.h>
#include <hip/hip_bf16.h>
#include <math.h>

#define C_DIM 96
#define HEADS 2
#define CH    48
#define HID   255
#define DD    16
#define HH    96
#define WW    96
#define NTOT  (DD * HH * WW)     // 147456
#define GRAM_CHUNKS 96
#define MBLK  2
#define NBLK  4

typedef __attribute__((ext_vector_type(16))) _Float16 v16h;
typedef __attribute__((ext_vector_type(8)))  _Float16 v8h;
typedef __attribute__((ext_vector_type(8)))  float    v8f;

// ---------------- BiasFree LayerNorm (channel axis) -> voxel-major f16 [N, 96]
__global__ __launch_bounds__(256)
void k_ln_t(const float* __restrict__ x, const float* __restrict__ w,
            _Float16* __restrict__ yt) {
  int n = blockIdx.x * 256 + threadIdx.x;
  if (n >= NTOT) return;
  float s = 0.f, s2 = 0.f;
  for (int c = 0; c < C_DIM; ++c) {
    float v = x[(size_t)c * NTOT + n];
    s += v; s2 += v * v;
  }
  float mu  = s * (1.0f / C_DIM);
  float var = fmaxf(s2 * (1.0f / C_DIM) - mu * mu, 0.f);
  float inv = rsqrtf(var + 1e-5f);
  for (int c8 = 0; c8 < C_DIM; c8 += 8) {
    v8h o;
#pragma unroll
    for (int j = 0; j < 8; ++j)
      o[j] = (_Float16)(x[(size_t)(c8 + j) * NTOT + n] * inv * w[c8 + j]);
    *(v8h*)(yt + (size_t)n * C_DIM + c8) = o;
  }
}

// ---------------- Register-blocked WMMA GEMM: out[M,N] = Wh[M,Kp] @ Bt[N,Kp]^T (+resid)
// One wave computes a (16*MBLK) x (16*NBLK) block: A fragments reused NBLK times,
// B fragments reused MBLK times -> 1.5 b128 loads per v_wmma (vs 4.0 unblocked).
__global__ __launch_bounds__(256)
void k_gemm_wmma(const _Float16* __restrict__ Wh,
                 const _Float16* __restrict__ Bt,
                 const float* __restrict__ resid,
                 float* __restrict__ outF,
                 _Float16* __restrict__ outH,
                 int Mpad, int Mreal, int Kp) {
  int wave = (int)((blockIdx.x * 256 + threadIdx.x) >> 5);
  int lane = threadIdx.x & 31;
  const int ntn = NTOT / (16 * NBLK);
  int mb = wave / ntn;
  int nb = wave - mb * ntn;
  if (mb >= Mpad / (16 * MBLK)) return;        // wave-uniform
  int lh = lane >> 4, lm = lane & 15;
  int m0 = mb * 16 * MBLK;
  int n0 = nb * 16 * NBLK;
  // A layout (f16 16x32): lanes 0-15: K 0-15 ; lanes 16-31: K 16-31
  const _Float16* aB0 = Wh + (size_t)(m0 + lm) * Kp + (lh ? 16 : 0);
  // B layout (f16 32x16): lanes 0-15: K 0-15 ; lanes 16-31: K 16-31 ; N = lane&15
  const _Float16* bB = Bt + (size_t)(n0 + lm) * Kp + (lh ? 16 : 0);
  v8f acc[MBLK][NBLK] = {};
  for (int kk = 0; kk < Kp; kk += 32) {
    v16h a[MBLK];
#pragma unroll
    for (int u = 0; u < MBLK; ++u) {
      const _Float16* ap = aB0 + (size_t)u * 16 * Kp + kk;
      v8h lo = *(const v8h*)(ap);
      v8h hi = *(const v8h*)(ap + 8);
      a[u] = __builtin_shufflevector(lo, hi, 0,1,2,3,4,5,6,7,8,9,10,11,12,13,14,15);
    }
#pragma unroll
    for (int t = 0; t < NBLK; ++t) {
      v16h b = *(const v16h*)(bB + (size_t)t * 16 * Kp + kk);
#pragma unroll
      for (int u = 0; u < MBLK; ++u)
        acc[u][t] = __builtin_amdgcn_wmma_f32_16x16x32_f16(false, a[u], false, b,
                                                           (short)0, acc[u][t], false, false);
    }
  }
#pragma unroll
  for (int u = 0; u < MBLK; ++u)
#pragma unroll
    for (int t = 0; t < NBLK; ++t)
#pragma unroll
      for (int r = 0; r < 8; ++r) {
        int m = m0 + u * 16 + r + (lh << 3);
        if (m >= Mreal) continue;
        size_t idx = (size_t)m * NTOT + (n0 + t * 16 + lm);
        float v = acc[u][t][r];
        if (resid) v += resid[idx];
        if (outF) outF[idx] = v; else outH[idx] = (_Float16)v;
      }
}

// ---------------- Depthwise 3x3x3 (replicate pad) for qkv; q,k -> [192,N] f16 ; v -> [N,96] f16
__global__ __launch_bounds__(256)
void k_dw_qkv(const _Float16* __restrict__ in, const float* __restrict__ wdw,
              _Float16* __restrict__ qk, _Float16* __restrict__ vt) {
  int gid = blockIdx.x * 256 + threadIdx.x;
  if (gid >= 3 * C_DIM * NTOT) return;
  int n = gid % NTOT;
  int c = gid / NTOT;
  int w = n % WW;
  int h = (n / WW) % HH;
  int d = n / (WW * HH);
  const _Float16* src = in + (size_t)c * NTOT;
  const float* wt = wdw + c * 27;
  float acc = 0.f;
  int t = 0;
#pragma unroll
  for (int dz = -1; dz <= 1; ++dz) {
    int dd = min(max(d + dz, 0), DD - 1);
#pragma unroll
    for (int dy = -1; dy <= 1; ++dy) {
      int hy = min(max(h + dy, 0), HH - 1);
#pragma unroll
      for (int dx = -1; dx <= 1; ++dx) {
        int wx = min(max(w + dx, 0), WW - 1);
        acc += (float)src[(size_t)(dd * HH + hy) * WW + wx] * wt[t++];
      }
    }
  }
  if (c < 2 * C_DIM) qk[(size_t)c * NTOT + n] = (_Float16)acc;
  else               vt[(size_t)n * C_DIM + (c - 2 * C_DIM)] = (_Float16)acc;
}

// ---------------- Split-K WMMA Gram: gram[h] += Q[48,N] @ K[48,N]^T (raw, un-normalized)
__global__ __launch_bounds__(256)
void k_gram_wmma(const _Float16* __restrict__ qk, float* __restrict__ gram) {
  int wave = (int)((blockIdx.x * 256 + threadIdx.x) >> 5);
  int lane = threadIdx.x & 31;
  int tile = wave / GRAM_CHUNKS;
  int chunk = wave - tile * GRAM_CHUNKS;
  if (tile >= 18) return;                      // wave-uniform
  int head = tile / 9;
  int rem = tile - head * 9;
  int ti = rem / 3, tj = rem - ti * 3;
  int lh = lane >> 4, lm = lane & 15;
  int qrow = head * CH + (ti << 4) + lm;
  int krow = C_DIM + head * CH + (tj << 4) + lm;
  const int steps = (NTOT / 32) / GRAM_CHUNKS; // 48
  int n0 = chunk * steps * 32;
  const _Float16* aB = qk + (size_t)qrow * NTOT + n0 + (lh ? 16 : 0);
  const _Float16* bB = qk + (size_t)krow * NTOT + n0 + (lh ? 16 : 0);
  v8f acc = {};
  for (int s = 0; s < steps; ++s) {
    int kk = s * 32;
    v8h alo = *(const v8h*)(aB + kk);
    v8h ahi = *(const v8h*)(aB + kk + 8);
    v16h a = __builtin_shufflevector(alo, ahi, 0,1,2,3,4,5,6,7,8,9,10,11,12,13,14,15);
    v16h b = *(const v16h*)(bB + kk);
    acc = __builtin_amdgcn_wmma_f32_16x16x32_f16(false, a, false, b,
                                                 (short)0, acc, false, false);
  }
#pragma unroll
  for (int r = 0; r < 8; ++r) {
    int i = (ti << 4) + r + (lh << 3);
    int j = (tj << 4) + lm;
    atomicAdd(&gram[head * CH * CH + i * CH + j], acc[r]);
  }
}

// ---------------- Row L2 norms of q,k (192 rows)
__global__ __launch_bounds__(256)
void k_rownorm(const _Float16* __restrict__ qk, float* __restrict__ norms) {
  __shared__ float sm[256];
  int row = blockIdx.x;
  float s = 0.f;
  for (int n = threadIdx.x; n < NTOT; n += 256) {
    float v = (float)qk[(size_t)row * NTOT + n];
    s += v * v;
  }
  sm[threadIdx.x] = s;
  __syncthreads();
  for (int off = 128; off > 0; off >>= 1) {
    if ((int)threadIdx.x < off) sm[threadIdx.x] += sm[threadIdx.x + off];
    __syncthreads();
  }
  if (threadIdx.x == 0) norms[row] = fmaxf(sqrtf(sm[0]), 1e-12f);
}

// ---------------- Softmax over 48, with normalization + temperature folded in
__global__ void k_softmax(const float* __restrict__ gram, const float* __restrict__ norms,
                          const float* __restrict__ temp, float* __restrict__ attn) {
  int idx = blockIdx.x * 64 + threadIdx.x;
  if (idx >= HEADS * CH) return;
  int h = idx / CH, i = idx - h * CH;
  float nq = norms[h * CH + i];
  float t = temp[h];
  float row[CH];
  float mx = -1e30f;
  for (int j = 0; j < CH; ++j) {
    float nk = norms[C_DIM + h * CH + j];
    float v = gram[(h * CH + i) * CH + j] / (nq * nk) * t;
    row[j] = v; mx = fmaxf(mx, v);
  }
  float sum = 0.f;
  for (int j = 0; j < CH; ++j) { float e = expf(row[j] - mx); row[j] = e; sum += e; }
  float inv = 1.f / sum;
  for (int j = 0; j < CH; ++j) attn[(h * CH + i) * CH + j] = row[j] * inv;
}

// ---------------- fused[96,96] = w_po @ blockdiag(attn)  (f16 out)
__global__ void k_fuse_po(const float* __restrict__ wpo, const float* __restrict__ attn,
                          _Float16* __restrict__ fh) {
  int idx = blockIdx.x * 256 + threadIdx.x;
  if (idx >= C_DIM * C_DIM) return;
  int m = idx / C_DIM, eg = idx - m * C_DIM;
  int h = eg / CH, e = eg - h * CH;
  float s = 0.f;
  for (int c = 0; c < CH; ++c)
    s += wpo[m * C_DIM + h * CH + c] * attn[(h * CH + c) * CH + e];
  fh[idx] = (_Float16)s;
}

// ---------------- FFN: depthwise conv on both halves + exact-GELU gating, fused.
// h1: [510, N] f16 channel-major -> gt: [N, 256] f16 voxel-major (col 255 zero pad)
__global__ __launch_bounds__(256)
void k_dw_gate(const _Float16* __restrict__ h1, const float* __restrict__ wdw,
               _Float16* __restrict__ gt) {
  int gid = blockIdx.x * 256 + threadIdx.x;
  if (gid >= 256 * NTOT) return;
  int n = gid % NTOT;
  int c = gid / NTOT;
  if (c == HID) { gt[(size_t)n * 256 + c] = (_Float16)0.f; return; }
  int w = n % WW;
  int h = (n / WW) % HH;
  int d = n / (WW * HH);
  const _Float16* s1 = h1 + (size_t)c * NTOT;
  const _Float16* s2 = h1 + (size_t)(c + HID) * NTOT;
  const float* w1 = wdw + c * 27;
  const float* w2 = wdw + (c + HID) * 27;
  float a = 0.f, b = 0.f;
  int t = 0;
#pragma unroll
  for (int dz = -1; dz <= 1; ++dz) {
    int dd = min(max(d + dz, 0), DD - 1);
#pragma unroll
    for (int dy = -1; dy <= 1; ++dy) {
      int hy = min(max(h + dy, 0), HH - 1);
#pragma unroll
      for (int dx = -1; dx <= 1; ++dx) {
        int wx = min(max(w + dx, 0), WW - 1);
        size_t off = (size_t)(dd * HH + hy) * WW + wx;
        a += (float)s1[off] * w1[t];
        b += (float)s2[off] * w2[t];
        ++t;
      }
    }
  }
  float g = 0.5f * a * (1.f + erff(a * 0.70710678118654752f)); // exact GELU
  gt[(size_t)n * 256 + c] = (_Float16)(g * b);
}

// ---------------- f32 -> f16 weight convert with zero padding
__global__ void k_cvt(const float* __restrict__ src, _Float16* __restrict__ dst,
                      int Ms, int Ks, int Mp, int Kp) {
  int idx = blockIdx.x * 256 + threadIdx.x;
  if (idx >= Mp * Kp) return;
  int m = idx / Kp, k = idx - m * Kp;
  dst[idx] = (m < Ms && k < Ks) ? (_Float16)src[m * Ks + k] : (_Float16)0.f;
}

__global__ void k_zero(float* __restrict__ p, int n) {
  int i = blockIdx.x * 256 + threadIdx.x;
  if (i < n) p[i] = 0.f;
}

extern "C" void kernel_launch(void* const* d_in, const int* in_sizes, int n_in,
                              void* d_out, int out_size, void* d_ws, size_t ws_size,
                              hipStream_t stream) {
  (void)in_sizes; (void)n_in; (void)out_size; (void)ws_size;
  const float* x     = (const float*)d_in[0];
  const float* ln1w  = (const float*)d_in[1];
  const float* temp  = (const float*)d_in[2];
  const float* wqkv  = (const float*)d_in[3];
  const float* wdw   = (const float*)d_in[4];
  const float* wpo   = (const float*)d_in[5];
  const float* ln2w  = (const float*)d_in[6];
  const float* win   = (const float*)d_in[7];
  const float* wffdw = (const float*)d_in[8];
  const float* wout  = (const float*)d_in[9];
  float* out = (float*)d_out;

  char* p = (char*)d_ws;
  auto carve = [&](size_t bytes) -> char* {
    char* r = p; p += (bytes + 255) & ~(size_t)255; return r;
  };
  _Float16* yt    = (_Float16*)carve((size_t)NTOT * C_DIM * 2);      // LN out (reused for LN2)
  _Float16* act   = (_Float16*)carve((size_t)512 * NTOT * 2);        // qkv f16, later h1 f16
  _Float16* qk    = (_Float16*)carve((size_t)2 * C_DIM * NTOT * 2);  // q,k f16 [192,N]
  _Float16* vt    = (_Float16*)carve((size_t)NTOT * C_DIM * 2);      // v voxel-major
  float*    x2    = (float*)carve((size_t)C_DIM * NTOT * 4);         // after attention residual
  float*    gram  = (float*)carve((size_t)HEADS * CH * CH * 4);
  float*    attn  = (float*)carve((size_t)HEADS * CH * CH * 4);
  float*    norms = (float*)carve((size_t)2 * C_DIM * 4);
  _Float16* wqkvh = (_Float16*)carve((size_t)3 * C_DIM * C_DIM * 2);
  _Float16* winh  = (_Float16*)carve((size_t)512 * C_DIM * 2);
  _Float16* wouth = (_Float16*)carve((size_t)C_DIM * 256 * 2);
  _Float16* fh    = (_Float16*)carve((size_t)C_DIM * C_DIM * 2);
  _Float16* gt    = qk;  // overlay: [N,256] f16 fits in qk+vt region, lifetimes disjoint

  const int TPB = 256;
  auto gemm_blocks = [](int Mpad) {
    int waves = (Mpad / (16 * MBLK)) * (NTOT / (16 * NBLK));
    return (waves + 7) / 8;
  };

  // weight conversions (f32 -> f16, zero-padded to tile multiples)
  k_cvt<<<(3 * C_DIM * C_DIM + TPB - 1) / TPB, TPB, 0, stream>>>(wqkv, wqkvh, 3 * C_DIM, C_DIM, 3 * C_DIM, C_DIM);
  k_cvt<<<(512 * C_DIM + TPB - 1) / TPB, TPB, 0, stream>>>(win, winh, 2 * HID, C_DIM, 512, C_DIM);
  k_cvt<<<(C_DIM * 256 + TPB - 1) / TPB, TPB, 0, stream>>>(wout, wouth, C_DIM, HID, C_DIM, 256);

  // ---- attention branch ----
  k_ln_t<<<(NTOT + TPB - 1) / TPB, TPB, 0, stream>>>(x, ln1w, yt);
  k_gemm_wmma<<<gemm_blocks(288), TPB, 0, stream>>>(wqkvh, yt, nullptr, nullptr, act, 288, 288, C_DIM);
  k_dw_qkv<<<(3 * C_DIM * NTOT + TPB - 1) / TPB, TPB, 0, stream>>>(act, wdw, qk, vt);
  k_zero<<<(HEADS * CH * CH + TPB - 1) / TPB, TPB, 0, stream>>>(gram, HEADS * CH * CH);
  k_rownorm<<<2 * C_DIM, TPB, 0, stream>>>(qk, norms);
  k_gram_wmma<<<(18 * GRAM_CHUNKS) / 8, TPB, 0, stream>>>(qk, gram);
  k_softmax<<<(HEADS * CH + 63) / 64, 64, 0, stream>>>(gram, norms, temp, attn);
  k_fuse_po<<<(C_DIM * C_DIM + TPB - 1) / TPB, TPB, 0, stream>>>(wpo, attn, fh);
  // x2 = x + (w_po@attn) @ v
  k_gemm_wmma<<<gemm_blocks(C_DIM), TPB, 0, stream>>>(fh, vt, x, x2, nullptr, C_DIM, C_DIM, C_DIM);

  // ---- GDFN feed-forward ----
  k_ln_t<<<(NTOT + TPB - 1) / TPB, TPB, 0, stream>>>(x2, ln2w, yt);
  k_gemm_wmma<<<gemm_blocks(512), TPB, 0, stream>>>(winh, yt, nullptr, nullptr, act, 512, 2 * HID, C_DIM);
  k_dw_gate<<<(256 * NTOT + TPB - 1) / TPB, TPB, 0, stream>>>(act, wffdw, gt);
  // out = x2 + w_out @ g
  k_gemm_wmma<<<gemm_blocks(C_DIM), TPB, 0, stream>>>(wouth, gt, x2, out, nullptr, C_DIM, C_DIM, 256);
}
